// MambaBlock_87935160418916
// MI455X (gfx1250) — matrix operand order repaired and checked by reading, not compiled, not run on hardware
//
#include <hip/hip_runtime.h>

typedef __bf16 bf16_t;
typedef __attribute__((ext_vector_type(16))) __bf16 v16bf;
typedef __attribute__((ext_vector_type(8)))  float  v8f;

constexpr int D_MODEL = 1024;
constexpr int D_STATE = 16;
constexpr int SEQ     = 2048;
constexpr int D_INNER = 2048;
constexpr int DT_RANK = 64;
constexpr int D_FF    = 4096;
constexpr int XD      = DT_RANK + 2 * D_STATE;   // 96
constexpr int MTOK    = 2 * SEQ;                 // 4096 tokens
constexpr float EPS   = 1e-6f;

union FragAB { v16bf v; unsigned int u[8]; };

// ---------------------------------------------------------------------------
// Generic bf16 WMMA GEMM: C[M,N] (f32) = A[M,K](bf16, row-major, lda)
//                                       x W[N,K](bf16, row-major, ldw)^T
// Block: 256 threads = 8 waves; each wave computes a 32x32 C tile
// (block tile 256Mx32N).  K must be a multiple of 64.
//
// The 32xK weight strip is shared by all 8 waves, so it is staged in LDS
// with CDNA5 async global->LDS copies (GLOBAL_LOAD_ASYNC_TO_LDS_B128,
// ASYNCcnt), double-buffered in 64-K chunks to overlap copy with WMMA.
// ---------------------------------------------------------------------------
__global__ __launch_bounds__(256)
void gemm_bf16_wmma(const bf16_t* __restrict__ A, const bf16_t* __restrict__ W,
                    float* __restrict__ C, int K, int lda, int ldw, int ldc) {
  __shared__ bf16_t sw[2][32 * 64];              // double-buffered W chunk, 8 KB

  const int tid   = threadIdx.x;
  const int lane  = tid & 31;
  const int wave  = tid >> 5;
  const int hf    = lane >> 4;                   // half-wave select
  const int l15   = lane & 15;
  const int tileN = blockIdx.x * 32;
  const int tileM = blockIdx.y * 256 + wave * 32;

  // Async staging: thread tid copies 16 bytes of the 32x64 chunk:
  // row n = tid/8, k-offset = (tid%8)*8 elements.
  const int srow  = tid >> 3;
  const int skoff = (tid & 7) * 8;
  const bf16_t* wsrc = W + (size_t)(tileN + srow) * ldw + skoff;
  const unsigned ldsd0 = (unsigned)(uintptr_t)&sw[0][srow * 64 + skoff];
  const unsigned ldsd1 = (unsigned)(uintptr_t)&sw[1][srow * 64 + skoff];

  auto issue_chunk = [&](int kc, int buf) {
    const unsigned long long ga = (unsigned long long)(uintptr_t)(wsrc + kc);
    const unsigned ld = buf ? ldsd1 : ldsd0;
    asm volatile("global_load_async_to_lds_b128 %0, %1, off"
                 :: "v"(ld), "v"(ga) : "memory");
  };

  issue_chunk(0, 0);

  v8f acc00 = {}, acc01 = {}, acc10 = {}, acc11 = {};

  const bf16_t* A0 = A + (size_t)(tileM + l15)      * lda;
  const bf16_t* A1 = A + (size_t)(tileM + 16 + l15) * lda;

  const int nchunks = K >> 6;
  for (int c = 0; c < nchunks; ++c) {
    const int buf = c & 1;
    if (c + 1 < nchunks) {
      issue_chunk((c + 1) << 6, buf ^ 1);        // prefetch next chunk
      asm volatile("s_wait_asynccnt 0x1" ::: "memory");   // chunk c landed (ours)
    } else {
      asm volatile("s_wait_asynccnt 0x0" ::: "memory");
    }
    __syncthreads();                             // all waves' pieces landed
    const bf16_t* wb = &sw[buf][0];

#pragma unroll
    for (int ks = 0; ks < 2; ++ks) {             // two 32-K WMMA steps per chunk
      const int k0g = (c << 6) + ks * 32;        // global k for A
      const int k0l = ks * 32;                   // chunk-local k for B
      FragAB a0, a1, b0, b1;
      // A fragment per ISA 16-bit 16x32 layout:
      // v=0..3: K = k0 + 8*hf + 2v ; v=4..7: +16
#pragma unroll
      for (int v = 0; v < 4; ++v) {
        const int ka = k0g + hf * 8 + 2 * v;
        a0.u[v]     = *(const unsigned int*)(A0 + ka);
        a0.u[v + 4] = *(const unsigned int*)(A0 + ka + 16);
        a1.u[v]     = *(const unsigned int*)(A1 + ka);
        a1.u[v + 4] = *(const unsigned int*)(A1 + ka + 16);
      }
      // B fragment (32x16): VGPR v holds K = k0 + 8*hf + 2(v&3) + 16*(v>>2), N = l15.
      // Per lane this is two 16-byte runs (K 8*hf+0..7 and +16) of one LDS-resident W row.
#pragma unroll
      for (int v = 0; v < 8; ++v) {
        const int kk = k0l + hf * 8 + 2 * (v & 3) + (v >> 2) * 16;
        b0.u[v] = *(const unsigned int*)(wb + l15 * 64 + kk);
        b1.u[v] = *(const unsigned int*)(wb + (l15 + 16) * 64 + kk);
      }
      acc00 = __builtin_amdgcn_wmma_f32_16x16x32_bf16(false, a0.v, false, b0.v, (short)0, acc00, false, false);
      acc01 = __builtin_amdgcn_wmma_f32_16x16x32_bf16(false, a0.v, false, b1.v, (short)0, acc01, false, false);
      acc10 = __builtin_amdgcn_wmma_f32_16x16x32_bf16(false, a1.v, false, b0.v, (short)0, acc10, false, false);
      acc11 = __builtin_amdgcn_wmma_f32_16x16x32_bf16(false, a1.v, false, b1.v, (short)0, acc11, false, false);
    }
    __syncthreads();                             // buffer reusable next iter
  }

  // C/D layout: VGPR i -> row (i + 8*hf), col l15 (both half-waves same cols)
#pragma unroll
  for (int i = 0; i < 8; ++i) {
    const size_t m0 = (size_t)tileM + i + 8 * hf;
    const int    n0 = tileN + l15;
    C[m0 * ldc + n0]             = acc00[i];
    C[m0 * ldc + n0 + 16]        = acc01[i];
    C[(m0 + 16) * ldc + n0]      = acc10[i];
    C[(m0 + 16) * ldc + n0 + 16] = acc11[i];
  }
}

// ---------------------------------------------------------------------------
// Elementwise helpers
// ---------------------------------------------------------------------------
__global__ __launch_bounds__(256)
void f32_to_bf16(const float* __restrict__ in, bf16_t* __restrict__ out, size_t n) {
  for (size_t i = blockIdx.x * 256ull + threadIdx.x; i < n; i += (size_t)gridDim.x * 256ull)
    out[i] = (bf16_t)in[i];
}

__global__ __launch_bounds__(256)
void silu_mul_to_bf16(const float* __restrict__ f1, const float* __restrict__ f3,
                      bf16_t* __restrict__ out, size_t n) {
  for (size_t i = blockIdx.x * 256ull + threadIdx.x; i < n; i += (size_t)gridDim.x * 256ull) {
    const float a = f1[i];
    out[i] = (bf16_t)((a / (1.f + __expf(-a))) * f3[i]);
  }
}

__global__ __launch_bounds__(256)
void add_f32(const float* __restrict__ a, const float* __restrict__ b,
             float* __restrict__ out, size_t n) {
  for (size_t i = blockIdx.x * 256ull + threadIdx.x; i < n; i += (size_t)gridDim.x * 256ull)
    out[i] = a[i] + b[i];
}

// ---------------------------------------------------------------------------
// RMSNorm (one block of 256 threads per token row, D_MODEL = 1024)
// ---------------------------------------------------------------------------
__global__ __launch_bounds__(256)
void rmsnorm_to_bf16(const float* __restrict__ x, const float* __restrict__ w,
                     bf16_t* __restrict__ out) {
  const size_t row = blockIdx.x;
  const float* xr = x + row * D_MODEL;
  float s = 0.f;
  float vals[4];
#pragma unroll
  for (int j = 0; j < 4; ++j) {
    const float v = xr[threadIdx.x + 256 * j];
    vals[j] = v; s += v * v;
  }
  __shared__ float red[256];
  red[threadIdx.x] = s; __syncthreads();
  for (int off = 128; off > 0; off >>= 1) {
    if (threadIdx.x < (unsigned)off) red[threadIdx.x] += red[threadIdx.x + off];
    __syncthreads();
  }
  const float inv = rsqrtf(red[0] / (float)D_MODEL + EPS);
#pragma unroll
  for (int j = 0; j < 4; ++j) {
    const int i = threadIdx.x + 256 * j;
    out[row * D_MODEL + i] = (bf16_t)(vals[j] * inv * w[i]);
  }
}

// residual add + RMSNorm, writes x2 (f32) and normalized bf16
__global__ __launch_bounds__(256)
void residual_rmsnorm_to_bf16(const float* __restrict__ x, const float* __restrict__ r,
                              const float* __restrict__ w, float* __restrict__ x2,
                              bf16_t* __restrict__ out) {
  const size_t row = blockIdx.x;
  const float* xr = x + row * D_MODEL;
  const float* rr = r + row * D_MODEL;
  float* x2r = x2 + row * D_MODEL;
  float s = 0.f;
  float vals[4];
#pragma unroll
  for (int j = 0; j < 4; ++j) {
    const int i = threadIdx.x + 256 * j;
    const float v = xr[i] + rr[i];
    vals[j] = v; x2r[i] = v; s += v * v;
  }
  __shared__ float red[256];
  red[threadIdx.x] = s; __syncthreads();
  for (int off = 128; off > 0; off >>= 1) {
    if (threadIdx.x < (unsigned)off) red[threadIdx.x] += red[threadIdx.x + off];
    __syncthreads();
  }
  const float inv = rsqrtf(red[0] / (float)D_MODEL + EPS);
#pragma unroll
  for (int j = 0; j < 4; ++j) {
    const int i = threadIdx.x + 256 * j;
    out[row * D_MODEL + i] = (bf16_t)(vals[j] * inv * w[i]);
  }
}

// ---------------------------------------------------------------------------
// Causal depthwise conv (k=4) over time + bias + SiLU.
// xz is (MTOK, 2*D_INNER); x-half is cols [0, D_INNER).
// ---------------------------------------------------------------------------
__global__ __launch_bounds__(256)
void conv_silu(const float* __restrict__ xz, const float* __restrict__ cw,
               const float* __restrict__ cb, float* __restrict__ xin_f,
               bf16_t* __restrict__ xin_b) {
  const size_t total = (size_t)MTOK * D_INNER;
  for (size_t idx = blockIdx.x * 256ull + threadIdx.x; idx < total;
       idx += (size_t)gridDim.x * 256ull) {
    const int d   = (int)(idx & (D_INNER - 1));
    const size_t row = idx >> 11;               // / D_INNER
    const int l   = (int)(row & (SEQ - 1));     // position within sequence
    float acc = cb[d];
#pragma unroll
    for (int j = 0; j < 4; ++j) {
      const int ls = l - 3 + j;
      if (ls >= 0)
        acc += cw[d * 4 + j] * xz[(row - (size_t)(3 - j)) * (2 * D_INNER) + d];
    }
    const float s = acc / (1.f + __expf(-acc));
    xin_f[idx] = s;
    xin_b[idx] = (bf16_t)s;
  }
}

// ---------------------------------------------------------------------------
// Selective scan. One thread per (batch, channel); 16-state in VGPRs.
// Fuses softplus(dt + bias), scan, +D*x skip, *silu(z), bf16 output.
// ---------------------------------------------------------------------------
__global__ __launch_bounds__(256)
void selective_scan(const float* __restrict__ dtg, const float* __restrict__ dt_b,
                    const float* __restrict__ xdbl, const float* __restrict__ xin,
                    const float* __restrict__ xz, const float* __restrict__ A_log,
                    const float* __restrict__ Dvec, bf16_t* __restrict__ ybf) {
  const int gid = blockIdx.x * 256 + threadIdx.x;   // 0 .. 4095
  const int b = gid / D_INNER;                      // same for whole block
  const int d = gid % D_INNER;

  float Arow[D_STATE], st[D_STATE];
#pragma unroll
  for (int n = 0; n < D_STATE; ++n) {
    Arow[n] = -__expf(A_log[d * D_STATE + n]);
    st[n] = 0.f;
  }
  const float bias = dt_b[d];
  const float Dd   = Dvec[d];

  __shared__ float sB[D_STATE], sC[D_STATE];

  for (int t = 0; t < SEQ; ++t) {
    const size_t row = (size_t)b * SEQ + t;
    __syncthreads();
    if (threadIdx.x < 32) {
      const float v = xdbl[row * XD + DT_RANK + threadIdx.x];
      if (threadIdx.x < 16) sB[threadIdx.x] = v;
      else                  sC[threadIdx.x - 16] = v;
    }
    __syncthreads();

    float dlt = dtg[row * D_INNER + d] + bias;
    dlt = (dlt > 20.f) ? dlt : __logf(1.f + __expf(dlt));   // softplus
    const float xt = xin[row * D_INNER + d];
    const float dx = dlt * xt;

    float y = 0.f;
#pragma unroll
    for (int n = 0; n < D_STATE; ++n) {
      const float dA = __expf(dlt * Arow[n]);
      st[n] = dA * st[n] + dx * sB[n];
      y += st[n] * sC[n];
    }
    const float yv = y + Dd * xt;
    const float z  = xz[row * (2 * D_INNER) + D_INNER + d];
    const float sz = z / (1.f + __expf(-z));
    ybf[row * D_INNER + d] = (bf16_t)(yv * sz);
  }
}

// ---------------------------------------------------------------------------
// Host-side orchestration
// ---------------------------------------------------------------------------
static inline unsigned ew_blocks(size_t n) {
  size_t b = (n + 255) / 256;
  return (unsigned)(b > 1048576 ? 1048576 : b);
}

extern "C" void kernel_launch(void* const* d_in, const int* in_sizes, int n_in,
                              void* d_out, int out_size, void* d_ws, size_t ws_size,
                              hipStream_t stream) {
  const float* x        = (const float*)d_in[0];
  const float* norm1_w  = (const float*)d_in[1];
  const float* norm2_w  = (const float*)d_in[2];
  const float* in_proj  = (const float*)d_in[3];
  const float* conv_w   = (const float*)d_in[4];
  const float* conv_b   = (const float*)d_in[5];
  const float* x_proj   = (const float*)d_in[6];
  const float* dt_proj  = (const float*)d_in[7];
  const float* dt_b     = (const float*)d_in[8];
  const float* A_log    = (const float*)d_in[9];
  const float* Dvec     = (const float*)d_in[10];
  const float* out_proj = (const float*)d_in[11];
  const float* w1       = (const float*)d_in[12];
  const float* w3       = (const float*)d_in[13];
  const float* w2       = (const float*)d_in[14];
  float* out = (float*)d_out;

  // scratch layout
  char* base = (char*)d_ws;
  size_t off = 0;
  auto alloc = [&](size_t bytes) -> char* {
    char* p = base + off;
    off += (bytes + 255) & ~(size_t)255;
    return p;
  };
  const size_t M = MTOK;
  bf16_t* h_bf    = (bf16_t*)alloc(M * D_MODEL * 2);       // rmsnorm1, reused for rmsnorm2
  bf16_t* win_bf  = (bf16_t*)alloc((size_t)2 * D_INNER * D_MODEL * 2);
  bf16_t* wxp_bf  = (bf16_t*)alloc((size_t)XD * D_INNER * 2);
  bf16_t* wdt_bf  = (bf16_t*)alloc((size_t)D_INNER * DT_RANK * 2);
  bf16_t* wout_bf = (bf16_t*)alloc((size_t)D_MODEL * D_INNER * 2);
  bf16_t* w1_bf   = (bf16_t*)alloc((size_t)D_FF * D_MODEL * 2);
  bf16_t* w3_bf   = (bf16_t*)alloc((size_t)D_FF * D_MODEL * 2);
  bf16_t* w2_bf   = (bf16_t*)alloc((size_t)D_MODEL * D_FF * 2);
  float*  xz      = (float*)alloc(M * 2 * D_INNER * 4);    // 64 MiB, aliased by f1 later
  float*  xin_f   = (float*)alloc(M * D_INNER * 4);        // 32 MiB \ contiguous: aliased
  float*  dtg     = (float*)alloc(M * D_INNER * 4);        // 32 MiB / by f3 (64 MiB)
  bf16_t* xin_b   = (bf16_t*)alloc(M * D_INNER * 2);
  float*  xdbl    = (float*)alloc(M * XD * 4);
  bf16_t* xdbl_bf = (bf16_t*)alloc(M * XD * 2);
  bf16_t* y_bf    = (bf16_t*)alloc(M * D_INNER * 2);       // 16 MiB \ contiguous: aliased
  float*  ssm_out = (float*)alloc(M * D_MODEL * 4);        // 16 MiB / by g_bf (32 MiB)
  float*  x2      = (float*)alloc(M * D_MODEL * 4);
  float*  mlp     = (float*)alloc(M * D_MODEL * 4);
  // aliases over dead buffers (sizes are exact multiples of 256, so regions abut)
  float*  f1   = xz;                // needs 64 MiB; xz dead after scan
  float*  f3   = xin_f;             // needs 64 MiB; xin_f+dtg dead after scan
  bf16_t* g_bf = (bf16_t*)y_bf;     // needs 32 MiB; y_bf+ssm_out dead after residual

  const dim3 B256(256);

  // 1. weights -> bf16
  f32_to_bf16<<<ew_blocks((size_t)2 * D_INNER * D_MODEL), B256, 0, stream>>>(in_proj, win_bf, (size_t)2 * D_INNER * D_MODEL);
  f32_to_bf16<<<ew_blocks((size_t)XD * D_INNER), B256, 0, stream>>>(x_proj, wxp_bf, (size_t)XD * D_INNER);
  f32_to_bf16<<<ew_blocks((size_t)D_INNER * DT_RANK), B256, 0, stream>>>(dt_proj, wdt_bf, (size_t)D_INNER * DT_RANK);
  f32_to_bf16<<<ew_blocks((size_t)D_MODEL * D_INNER), B256, 0, stream>>>(out_proj, wout_bf, (size_t)D_MODEL * D_INNER);
  f32_to_bf16<<<ew_blocks((size_t)D_FF * D_MODEL), B256, 0, stream>>>(w1, w1_bf, (size_t)D_FF * D_MODEL);
  f32_to_bf16<<<ew_blocks((size_t)D_FF * D_MODEL), B256, 0, stream>>>(w3, w3_bf, (size_t)D_FF * D_MODEL);
  f32_to_bf16<<<ew_blocks((size_t)D_MODEL * D_FF), B256, 0, stream>>>(w2, w2_bf, (size_t)D_MODEL * D_FF);

  // 2. h = rmsnorm(x, norm1_w) -> bf16
  rmsnorm_to_bf16<<<M, B256, 0, stream>>>(x, norm1_w, h_bf);

  // 3. xz = h @ in_proj^T   (M=4096, N=4096, K=1024)
  gemm_bf16_wmma<<<dim3(2 * D_INNER / 32, M / 256), B256, 0, stream>>>(
      h_bf, win_bf, xz, D_MODEL, D_MODEL, D_MODEL, 2 * D_INNER);

  // 4. causal depthwise conv + SiLU
  conv_silu<<<ew_blocks(M * D_INNER), B256, 0, stream>>>(xz, conv_w, conv_b, xin_f, xin_b);

  // 5. x_dbl = xin @ x_proj^T   (N=96, K=2048)
  gemm_bf16_wmma<<<dim3(XD / 32, M / 256), B256, 0, stream>>>(
      xin_b, wxp_bf, xdbl, D_INNER, D_INNER, D_INNER, XD);
  f32_to_bf16<<<ew_blocks(M * XD), B256, 0, stream>>>(xdbl, xdbl_bf, M * XD);

  // 6. dt_raw = x_dbl[:, :64] @ dt_proj^T   (N=2048, K=64, lda=96)
  gemm_bf16_wmma<<<dim3(D_INNER / 32, M / 256), B256, 0, stream>>>(
      xdbl_bf, wdt_bf, dtg, DT_RANK, XD, DT_RANK, D_INNER);

  // 7. fused softplus + selective scan + skip + silu(z) gate -> y_bf
  selective_scan<<<dim3((2 * D_INNER) / 256), B256, 0, stream>>>(   // 16 blocks
      dtg, dt_b, xdbl, xin_f, xz, A_log, Dvec, y_bf);

  // 8. ssm_out = y_gated @ out_proj^T   (N=1024, K=2048)
  gemm_bf16_wmma<<<dim3(D_MODEL / 32, M / 256), B256, 0, stream>>>(
      y_bf, wout_bf, ssm_out, D_INNER, D_INNER, D_INNER, D_MODEL);

  // 9. x2 = x + ssm_out ; h2 = rmsnorm(x2, norm2_w) -> bf16 (reuse h_bf)
  residual_rmsnorm_to_bf16<<<M, B256, 0, stream>>>(x, ssm_out, norm2_w, x2, h_bf);

  // 10. MLP
  gemm_bf16_wmma<<<dim3(D_FF / 32, M / 256), B256, 0, stream>>>(
      h_bf, w1_bf, f1, D_MODEL, D_MODEL, D_MODEL, D_FF);
  gemm_bf16_wmma<<<dim3(D_FF / 32, M / 256), B256, 0, stream>>>(
      h_bf, w3_bf, f3, D_MODEL, D_MODEL, D_MODEL, D_FF);
  silu_mul_to_bf16<<<ew_blocks(M * D_FF), B256, 0, stream>>>(f1, f3, g_bf, M * D_FF);
  gemm_bf16_wmma<<<dim3(D_MODEL / 32, M / 256), B256, 0, stream>>>(
      g_bf, w2_bf, mlp, D_FF, D_FF, D_FF, D_MODEL);

  // 11. out = x2 + mlp
  add_f32<<<ew_blocks(M * D_MODEL), B256, 0, stream>>>(x2, mlp, out, M * D_MODEL);
}
